// GATConv_55645596287279
// MI455X (gfx1250) — compile-verified, matrix operand order checked by reading, not run on hardware
//
#include <hip/hip_runtime.h>
#include <stdint.h>

#define F_IN      128
#define HC        128      // HEADS * C_OUT
#define HEADS     4
#define C_OUT     32
#define NEG_SLOPE 0.2f

typedef __attribute__((ext_vector_type(16))) __bf16         v16bf;
typedef __attribute__((ext_vector_type(8)))  float          v8f;
typedef __attribute__((ext_vector_type(8)))  unsigned short ushort8;

union ABTile {            // 32 bytes: two b128 loads -> one WMMA operand
  ushort8 u[2];
  v16bf   b;
};

__device__ __forceinline__ unsigned short f32_to_bf16_rne(float f) {
  unsigned u = __float_as_uint(f);
  unsigned r = u + 0x7FFFu + ((u >> 16) & 1u);   // round-to-nearest-even
  return (unsigned short)(r >> 16);
}

// Order-preserving encoding so atomicMax(u32) implements float max.
__device__ __forceinline__ unsigned enc_ord(float f) {
  unsigned u = __float_as_uint(f);
  return (u & 0x80000000u) ? ~u : (u | 0x80000000u);
}
__device__ __forceinline__ float dec_ord(unsigned u) {
  return (u & 0x80000000u) ? __uint_as_float(u & 0x7FFFFFFFu)
                           : __uint_as_float(~u);
}

__device__ __forceinline__ float leaky(float f) {
  return f >= 0.f ? f : NEG_SLOPE * f;
}

// ---------------- init: out = bias broadcast; segment max/sum reset ----------
__global__ void gat_init_out(float* __restrict__ out,
                             const float* __restrict__ bias, int n) {
  int i = blockIdx.x * blockDim.x + threadIdx.x;
  if (i < n * HC) out[i] = bias[i & (HC - 1)];
}

__global__ void gat_init_seg(unsigned* __restrict__ m, float* __restrict__ s,
                             int n) {
  int i = blockIdx.x * blockDim.x + threadIdx.x;
  if (i < n * HEADS) { m[i] = 0u; s[i] = 0.f; }   // 0 < enc(-inf) = 0x007FFFFF
}

// ---------------- edge_index_full = [edges | self-loops], written to d_out ---
__global__ void gat_write_edges(const int* __restrict__ edge_index,
                                int* __restrict__ eif, int e, int etot) {
  int i = blockIdx.x * blockDim.x + threadIdx.x;
  if (i >= etot) return;
  int s = (i < e) ? edge_index[i]     : (i - e);
  int d = (i < e) ? edge_index[e + i] : (i - e);
  eif[i] = s;
  eif[etot + i] = d;
}

// ---------------- fp32 -> bf16 casts (x, and W transposed for B-operand) -----
__global__ void gat_cast_x(const float* __restrict__ x,
                           unsigned short* __restrict__ xb, int count) {
  int i = blockIdx.x * blockDim.x + threadIdx.x;
  if (i < count) xb[i] = f32_to_bf16_rne(x[i]);
}

__global__ void gat_cast_wt(const float* __restrict__ W,
                            unsigned short* __restrict__ wtb) {
  int i = blockIdx.x * blockDim.x + threadIdx.x;  // i = nOut*F_IN + k
  if (i >= F_IN * HC) return;
  int nOut = i >> 7, k = i & 127;
  wtb[i] = f32_to_bf16_rne(W[(size_t)k * HC + nOut]);
}

// ---------------- h = x @ W via v_wmma_f32_16x16x32_bf16 ---------------------
// One block = 16-row slab of h. 8 waves; wave w owns the 16-col tile w.
// A layout (ISA 7.12.2, 16-bit A 16x32): lane&15 = row, lane>>4 selects the
// K half; VGPRs 0-3 = 8 contiguous K at halfk, VGPRs 4-7 = 8 contiguous K at
// halfk+16  ->  two b128 loads per k-step.  B symmetric on column-major W (Wt).
// Full blocks (r0+16 <= n) take a branch-free store path; only the tail block
// pays per-row exec-mask guards.
__global__ void gat_gemm_wmma(const unsigned short* __restrict__ xb,
                              const unsigned short* __restrict__ wtb,
                              float* __restrict__ h, int n_nodes) {
  const int  lane  = threadIdx.x & 31;
  const int  wave  = threadIdx.x >> 5;           // 0..7 => col tile
  const int  r0    = blockIdx.x << 4;            // 16 rows / block
  const bool full  = (r0 + 16) <= n_nodes;
  const int  halfk = (lane >> 4) << 3;           // 0 or 8
  const int  arow  = full ? (r0 + (lane & 15))
                          : min(r0 + (lane & 15), n_nodes - 1);
  const int  bcol  = (wave << 4) + (lane & 15);

  const unsigned short* __restrict__ ap = xb  + (size_t)arow * F_IN;
  const unsigned short* __restrict__ bp = wtb + (size_t)bcol * F_IN;

  v8f acc = {};
#pragma unroll
  for (int kt = 0; kt < 4; ++kt) {               // K = 128 = 4 x 32
    const int k0 = kt * 32 + halfk;
    ABTile a, b;
    a.u[0] = *(const ushort8*)(ap + k0);
    a.u[1] = *(const ushort8*)(ap + k0 + 16);
    b.u[0] = *(const ushort8*)(bp + k0);
    b.u[1] = *(const ushort8*)(bp + k0 + 16);
    acc = __builtin_amdgcn_wmma_f32_16x16x32_bf16(
        /*neg_a=*/false, a.b, /*neg_b=*/false, b.b,
        /*c_mod=*/(short)0, acc, /*reuse_a=*/false, /*reuse_b=*/false);
  }

  // C/D layout: VGPR r -> M = (lane>>4)*8 + r, N = lane&15
  const int nn    = lane & 15;
  const int mbase = (lane >> 4) << 3;
  const int c0    = wave << 4;
  float* __restrict__ hout = h + (size_t)(r0 + mbase) * HC + c0 + nn;
  if (full) {
#pragma unroll
    for (int r = 0; r < 8; ++r) hout[(size_t)r * HC] = acc[r];
  } else {
#pragma unroll
    for (int r = 0; r < 8; ++r)
      if (r0 + mbase + r < n_nodes) hout[(size_t)r * HC] = acc[r];
  }
}

// ---------------- a_src[n,h], a_dst[n,h] -------------------------------------
__global__ void gat_att(const float* __restrict__ h,
                        const float* __restrict__ att_src,
                        const float* __restrict__ att_dst,
                        float* __restrict__ asrc, float* __restrict__ adst,
                        int n_nodes) {
  int i = blockIdx.x * blockDim.x + threadIdx.x;      // (node, head)
  if (i >= n_nodes * HEADS) return;
  int head = i & (HEADS - 1);
  int node = i >> 2;
  const float* __restrict__ hp = h + (size_t)node * HC + head * C_OUT;
  const float* __restrict__ as = att_src + head * C_OUT;
  const float* __restrict__ ad = att_dst + head * C_OUT;
  float s0 = 0.f, s1 = 0.f;
#pragma unroll
  for (int c = 0; c < C_OUT; ++c) { s0 += hp[c] * as[c]; s1 += hp[c] * ad[c]; }
  asrc[i] = s0;
  adst[i] = s1;
}

// ---------------- segment max over dst (atomicMax on ordered-uint) -----------
__global__ void gat_edge_max(const int* __restrict__ esrc,
                             const int* __restrict__ edst,
                             const float* __restrict__ asrc,
                             const float* __restrict__ adst,
                             unsigned* __restrict__ m, int etot) {
  int e = blockIdx.x * blockDim.x + threadIdx.x;
  if (e >= etot) return;
  int s = esrc[e], d = edst[e];
  const float4 av = *(const float4*)(asrc + 4 * (size_t)s);
  const float4 dv = *(const float4*)(adst + 4 * (size_t)d);
  float el[HEADS] = {leaky(av.x + dv.x), leaky(av.y + dv.y),
                     leaky(av.z + dv.z), leaky(av.w + dv.w)};
#pragma unroll
  for (int hh = 0; hh < HEADS; ++hh)
    atomicMax(&m[(size_t)d * HEADS + hh], enc_ord(el[hh]));
}

// ---------------- e = exp(el - m[dst]); segment sum; stage e in alpha --------
__global__ void gat_edge_exp(const int* __restrict__ esrc,
                             const int* __restrict__ edst,
                             const float* __restrict__ asrc,
                             const float* __restrict__ adst,
                             const unsigned* __restrict__ m,
                             float* __restrict__ ssum,
                             float* __restrict__ alpha, int etot) {
  int e = blockIdx.x * blockDim.x + threadIdx.x;
  if (e >= etot) return;
  int s = esrc[e], d = edst[e];
  const float4 av = *(const float4*)(asrc + 4 * (size_t)s);
  const float4 dv = *(const float4*)(adst + 4 * (size_t)d);
  float el[HEADS] = {leaky(av.x + dv.x), leaky(av.y + dv.y),
                     leaky(av.z + dv.z), leaky(av.w + dv.w)};
#pragma unroll
  for (int hh = 0; hh < HEADS; ++hh) {
    float ex = __expf(el[hh] - dec_ord(m[(size_t)d * HEADS + hh]));
    atomicAdd(&ssum[(size_t)d * HEADS + hh], ex);
    alpha[(size_t)e * HEADS + hh] = ex;
  }
}

// ---------------- alpha = e / (sum + eps) ------------------------------------
__global__ void gat_edge_norm(const int* __restrict__ edst,
                              const float* __restrict__ ssum,
                              float* __restrict__ alpha, int etot) {
  int i = blockIdx.x * blockDim.x + threadIdx.x;     // (edge, head)
  if (i >= etot * HEADS) return;
  int e = i >> 2, hh = i & (HEADS - 1);
  int d = edst[e];
  alpha[i] = alpha[i] / (ssum[(size_t)d * HEADS + hh] + 1e-16f);
}

// ---------------- out[dst] += h[src] * alpha  (feature scatter) --------------
__global__ void gat_scatter(const int* __restrict__ esrc,
                            const int* __restrict__ edst,
                            const float* __restrict__ h,
                            const float* __restrict__ alpha,
                            float* __restrict__ out, int etot) {
  long long i = (long long)blockIdx.x * blockDim.x + threadIdx.x;
  if (i >= (long long)etot * HC) return;
  int e = (int)(i >> 7);
  int f = (int)(i & 127);
  int hh = f >> 5;
  float a = alpha[(size_t)e * HEADS + hh];
  atomicAdd(&out[(size_t)edst[e] * HC + f], h[(size_t)esrc[e] * HC + f] * a);
}

// =============================================================================
extern "C" void kernel_launch(void* const* d_in, const int* in_sizes, int n_in,
                              void* d_out, int out_size, void* d_ws,
                              size_t ws_size, hipStream_t stream) {
  const float* x          = (const float*)d_in[0];
  const int*   edge_index = (const int*)d_in[1];
  const float* W          = (const float*)d_in[2];
  const float* att_src    = (const float*)d_in[3];
  const float* att_dst    = (const float*)d_in[4];
  const float* bias       = (const float*)d_in[5];

  const int n    = in_sizes[0] / F_IN;
  const int e    = in_sizes[1] / 2;
  const int etot = e + n;

  // ---- workspace carve-up (256B aligned) ----
  char*  ws  = (char*)d_ws;
  size_t off = 0;
  auto carve = [&](size_t bytes) -> void* {
    void* p = ws + off;
    off = (off + bytes + 255) & ~(size_t)255;
    return p;
  };
  float*          h    = (float*)carve((size_t)n * HC * sizeof(float));
  unsigned short* xb   = (unsigned short*)carve((size_t)n * F_IN * 2);
  unsigned short* wtb  = (unsigned short*)carve((size_t)F_IN * HC * 2);
  float*          as   = (float*)carve((size_t)n * HEADS * sizeof(float));
  float*          ad   = (float*)carve((size_t)n * HEADS * sizeof(float));
  unsigned*       m    = (unsigned*)carve((size_t)n * HEADS * sizeof(unsigned));
  float*          ssum = (float*)carve((size_t)n * HEADS * sizeof(float));

  // ---- d_out carve-up: [out fp32 | edge_index_full i32 | alpha fp32] ----
  float* out   = (float*)d_out;
  int*   eif   = (int*)d_out + (size_t)n * HC;
  float* alpha = (float*)d_out + (size_t)n * HC + (size_t)2 * etot;

  const int T = 256;
  auto nb = [](long long work, int t) { return (unsigned)((work + t - 1) / t); };

  gat_init_out  <<<nb((long long)n * HC, T), T, 0, stream>>>(out, bias, n);
  gat_init_seg  <<<nb((long long)n * HEADS, T), T, 0, stream>>>(m, ssum, n);
  gat_write_edges<<<nb(etot, T), T, 0, stream>>>(edge_index, eif, e, etot);

  gat_cast_x    <<<nb((long long)n * F_IN, T), T, 0, stream>>>(x, xb, n * F_IN);
  gat_cast_wt   <<<nb(F_IN * HC, T), T, 0, stream>>>(W, wtb);

  gat_gemm_wmma <<<nb(n, 16), 256, 0, stream>>>(xb, wtb, h, n);   // 16 rows/blk

  gat_att       <<<nb((long long)n * HEADS, T), T, 0, stream>>>(h, att_src,
                                                                att_dst, as, ad, n);

  const int* esrc = eif;
  const int* edst = eif + etot;
  gat_edge_max  <<<nb(etot, T), T, 0, stream>>>(esrc, edst, as, ad, m, etot);
  gat_edge_exp  <<<nb(etot, T), T, 0, stream>>>(esrc, edst, as, ad, m, ssum,
                                                alpha, etot);
  gat_edge_norm <<<nb((long long)etot * HEADS, T), T, 0, stream>>>(edst, ssum,
                                                                   alpha, etot);
  gat_scatter   <<<nb((long long)etot * HC, T), T, 0, stream>>>(esrc, edst, h,
                                                                alpha, out, etot);
}
